// TextLoss_54125177864480
// MI455X (gfx1250) — matrix operand order repaired and checked by reading, not compiled, not run on hardware
//
#include <hip/hip_runtime.h>
#include <hip/hip_bf16.h>

typedef __attribute__((ext_vector_type(16))) _Float16 v16h;
typedef __attribute__((ext_vector_type(8)))  _Float16 v8h;
typedef __attribute__((ext_vector_type(8)))  float    v8f;

#define BB    8
#define HH    512
#define WWID  512
#define HW    (HH*WWID)
#define NPIX  (BB*HW)
#define NBINS 2048
#define BPB   32          // blocks per batch for per-batch kernels
#define KMAXI 8

// ---- workspace layout (float index) ----
enum {
  OFF_HIST   = 0,      // 2048 u32 (hist of neg CE)
  OFF_MU_ACC = 2048,   // 2048 f32: [b][16 M][16 N]; N=0..7 sum(e), N=8 counts
  OFF_PULL   = 4096,   // 64 f32: pull_acc[b][k]
  OFF_SCAL   = 4160,   // 32 scalar slots
  OFF_PUSHB  = 4192,   // 8 f32
  OFF_KB     = 4200,   // 8 f32
  OFF_MU     = 4208,   // 512 f32: mu[b][k][c]
  OFF_CNT    = 4720,   // 64 f32: counts[b][k]
  WS_FLOATS  = 8192
};
enum { S_NPOS=0,S_NNEGAV,S_N1,S_N2,S_HAS2,S_POSSUM,S_NEGTOT,S_TCLSUM,
       S_RADSUM,S_SINSUM,S_COSSUM,S_TBIN,S_NEEDAT,S_SUMABOVE,S_SUMAT,S_CNTAT };

__device__ __forceinline__ float ntn(float x) {
  if (isnan(x)) return 0.0f;
  if (isinf(x)) return x > 0.0f ? 1e5f : -1e5f;
  return x;
}
__device__ __forceinline__ float ce2(float a, float b, int t) {
  float m = fmaxf(a, b);
  float lse = m + logf(expf(a - m) + expf(b - m));
  return lse - (t ? b : a);
}
__device__ __forceinline__ float sl1(float d) {
  d = fabsf(d);
  return d < 1.0f ? 0.5f * d * d : d - 0.5f;
}
__device__ __forceinline__ int binOf(float ce) {
  int b = (int)(ce * 64.0f);           // bins over [0, 32)
  return b < 0 ? 0 : (b > NBINS - 1 ? NBINS - 1 : b);
}
__device__ __forceinline__ float waveSum(float v) {
  #pragma unroll
  for (int o = 16; o > 0; o >>= 1) v += __shfl_down(v, o, 32);
  return v;
}

// ---------------- kernel 0: zero workspace ----------------
__global__ void k_zero(float* W, int n) {
  for (int i = blockIdx.x * blockDim.x + threadIdx.x; i < n; i += gridDim.x * blockDim.x)
    W[i] = 0.0f;
}

// ---------------- kernel 1: elementwise pass (CE/smoothL1 sums + neg-CE histogram) ----
__global__ void k_pass1(const float* __restrict__ inp, const int* __restrict__ trm_,
                        const int* __restrict__ tclm_, const float* __restrict__ sm_,
                        const float* __restrict__ cm_, const float* __restrict__ rm_,
                        const int* __restrict__ tm_, float* W) {
  __shared__ unsigned histS[NBINS];
  for (int i = threadIdx.x; i < NBINS; i += blockDim.x) histS[i] = 0u;
  __syncthreads();

  float n_pos=0,n_neg=0,n1=0,n2=0,has2=0,pos_sum=0,neg_tot=0,tcl_sum=0,rad_sum=0,sin_sum=0,cos_sum=0;

  for (size_t i = (size_t)blockIdx.x * blockDim.x + threadIdx.x; i < (size_t)NPIX;
       i += (size_t)gridDim.x * blockDim.x) {
    size_t b = i / HW, hw = i % HW;     // HW = 2^18, pure shifts
    const float* ip = inp + b * 7 * (size_t)HW + hw;
    float l0 = ntn(ip[0]);            float l1 = ntn(ip[(size_t)HW]);
    float l2 = ntn(ip[2*(size_t)HW]); float l3 = ntn(ip[3*(size_t)HW]);
    float sp = ntn(ip[4*(size_t)HW]); float cp = ntn(ip[5*(size_t)HW]);
    float rp = ip[6*(size_t)HW];
    int trm = trm_[i], tclm = tclm_[i], tm = tm_[i];

    float ce_tr = ce2(l0, l1, trm);
    if (trm * tm > 0)       { n_pos += 1.0f; pos_sum += ce_tr; }
    if ((1 - trm) * tm > 0) { n_neg += 1.0f; neg_tot += ce_tr;
                              atomicAdd(&histS[binOf(ce_tr)], 1u); }
    if (tm * trm > 0)       { n1 += 1.0f; tcl_sum += ce2(l2, l3, tclm); }
    if (tclm > 0) {
      n2 += 1.0f;
      rad_sum += sl1(rp / (rm_[i] + 1e-6f) - 1.0f);
      float scale = sqrtf(1.0f / (sp * sp + cp * cp + 1e-6f));
      sin_sum += sl1(sp * scale - ntn(sm_[i]));
      cos_sum += sl1(cp * scale - ntn(cm_[i]));
    }
    if (tm * tclm > 0) has2 += 1.0f;
  }

  float vals[11]  = {n_pos,n_neg,n1,n2,has2,pos_sum,neg_tot,tcl_sum,rad_sum,sin_sum,cos_sum};
  const int slot[11] = {S_NPOS,S_NNEGAV,S_N1,S_N2,S_HAS2,S_POSSUM,S_NEGTOT,S_TCLSUM,
                        S_RADSUM,S_SINSUM,S_COSSUM};
  int lane = threadIdx.x & 31;
  #pragma unroll
  for (int j = 0; j < 11; ++j) {
    float r = waveSum(vals[j]);
    if (lane == 0) atomicAdd(&W[OFF_SCAL + slot[j]], r);
  }
  __syncthreads();
  for (int i = threadIdx.x; i < NBINS; i += blockDim.x)
    if (histS[i]) atomicAdd((unsigned*)W + OFF_HIST + i, histS[i]);
}

// ---------------- kernel 2: OHEM threshold select (scan histogram top-down) -------
__global__ void k_select(float* W) {
  if (threadIdx.x != 0 || blockIdx.x != 0) return;
  int n_pos = (int)(W[OFF_SCAL + S_NPOS] + 0.5f);
  int n_av  = (int)(W[OFF_SCAL + S_NNEGAV] + 0.5f);
  int n_neg = n_pos > 0 ? min(n_av, (int)(3.0f * (float)n_pos)) : 100;
  const unsigned* hist = (const unsigned*)W + OFF_HIST;
  long long acc = 0; int tbin = -1; int need = 0;
  for (int bn = NBINS - 1; bn >= 0; --bn) {
    long long prev = acc; acc += hist[bn];
    if (acc >= (long long)n_neg) { tbin = bn; need = (int)(n_neg - prev); break; }
  }
  W[OFF_SCAL + S_TBIN]   = (float)tbin;
  W[OFF_SCAL + S_NEEDAT] = (float)need;
}

// ---------------- kernel 3: second pass, sum negative CE above threshold ----------
__global__ void k_pass1c(const float* __restrict__ inp, const int* __restrict__ trm_,
                         const int* __restrict__ tm_, float* W) {
  int tbin = (int)W[OFF_SCAL + S_TBIN];
  float above = 0, at = 0, cntat = 0;
  for (size_t i = (size_t)blockIdx.x * blockDim.x + threadIdx.x; i < (size_t)NPIX;
       i += (size_t)gridDim.x * blockDim.x) {
    int trm = trm_[i], tm = tm_[i];
    if ((1 - trm) * tm > 0) {
      size_t b = i / HW, hw = i % HW;
      const float* ip = inp + b * 7 * (size_t)HW + hw;
      float ce = ce2(ntn(ip[0]), ntn(ip[(size_t)HW]), trm);
      int bn = binOf(ce);
      if (bn > tbin)        above += ce;
      else if (bn == tbin)  { at += ce; cntat += 1.0f; }
    }
  }
  int lane = threadIdx.x & 31;
  float r;
  r = waveSum(above); if (lane == 0) atomicAdd(&W[OFF_SCAL + S_SUMABOVE], r);
  r = waveSum(at);    if (lane == 0) atomicAdd(&W[OFF_SCAL + S_SUMAT], r);
  r = waveSum(cntat); if (lane == 0) atomicAdd(&W[OFF_SCAL + S_CNTAT], r);
}

// ---------------- kernel 4: WMMA accumulation of mu-sums and counts ----------------
// D = A(onehot 16x32 f16) x B(e 32x16 f16) + C, accumulated over K chunks of 32 px.
// B column N=c (c<8) carries embedding channel c; column N=8 carries 1.0 -> counts.
// Staging is wave-private: labels/masks via async global->LDS copies (ASYNCcnt),
// embeddings via VGPR cvt + ds_store into a transposed f16 tile (DScnt). No
// workgroup barriers; wave-level s_wait_{asynccnt,dscnt} only. Control flow around
// the WMMA is uniform (EXEC all ones).
__global__ void k_disc_mu(const float* __restrict__ emb, const int* __restrict__ lab,
                          const int* __restrict__ msk, float* W) {
  __shared__ __align__(16) _Float16 eT[8][16][32];   // [wave][col N][K]
  __shared__ __align__(16) int      labS[8][32];     // [wave][K]
  __shared__ __align__(16) int      mskS[8][32];
  const int wave = threadIdx.x >> 5;
  const int lane = threadIdx.x & 31;
  const int half = lane >> 4;
  const int l15  = lane & 15;
  const int b    = blockIdx.x / BPB;
  const int blk  = blockIdx.x % BPB;
  const float* eb = emb + (size_t)b * 8 * HW;
  const int*   lb = lab + (size_t)b * HW;
  const int*   mb = msk + (size_t)b * HW;

  // prefill constant B columns: N=8 -> 1.0 (counts), N=9..15 -> 0
  #pragma unroll
  for (int c = 8; c < 16; ++c)
    eT[wave][c][lane] = (c == 8) ? (_Float16)1.0f : (_Float16)0.0f;

  const unsigned ldsL = (unsigned)(size_t)&labS[wave][lane];
  const unsigned ldsM = (unsigned)(size_t)&mskS[wave][lane];

  v8f acc = {};
  for (int base = blk * 256 + wave * 32; base < HW; base += BPB * 256) {
    int pix = base + lane;

    // async copy label & mask words straight into LDS (tracked by ASYNCcnt)
    unsigned long long gaL = (unsigned long long)(size_t)(lb + pix);
    unsigned long long gaM = (unsigned long long)(size_t)(mb + pix);
    asm volatile("global_load_async_to_lds_b32 %0, %1, off"
                 :: "v"(ldsL), "v"(gaL) : "memory");
    asm volatile("global_load_async_to_lds_b32 %0, %1, off"
                 :: "v"(ldsM), "v"(gaM) : "memory");

    // prefetch next chunk (global_prefetch_b8)
    if (base + BPB * 256 < HW) {
      __builtin_prefetch(eb + pix + BPB * 256, 0, 1);
      __builtin_prefetch(lb + pix + BPB * 256, 0, 1);
    }

    // stage embedding chunk transposed as f16: eT[col=c][K=lane]
    #pragma unroll
    for (int c = 0; c < 8; ++c)
      eT[wave][c][lane] = (_Float16)eb[(size_t)c * HW + pix];

    // wave-local fences: all async copies + LDS stores visible to this wave
    asm volatile("s_wait_asynccnt 0" ::: "memory");
    asm volatile("s_wait_dscnt 0"    ::: "memory");

    // ---- A fragment (16-bit A 16x32): lane half h, row M=l15
    //      elem i<8  -> K = 8h + i      ; elem i>=8 -> K = 16 + 8h + (i-8)
    int4 la0 = *(const int4*)&labS[wave][8*half];
    int4 la1 = *(const int4*)&labS[wave][8*half + 4];
    int4 la2 = *(const int4*)&labS[wave][16 + 8*half];
    int4 la3 = *(const int4*)&labS[wave][16 + 8*half + 4];
    int4 ma0 = *(const int4*)&mskS[wave][8*half];
    int4 ma1 = *(const int4*)&mskS[wave][8*half + 4];
    int4 ma2 = *(const int4*)&mskS[wave][16 + 8*half];
    int4 ma3 = *(const int4*)&mskS[wave][16 + 8*half + 4];
    int li[16] = {la0.x,la0.y,la0.z,la0.w, la1.x,la1.y,la1.z,la1.w,
                  la2.x,la2.y,la2.z,la2.w, la3.x,la3.y,la3.z,la3.w};
    int mi[16] = {ma0.x,ma0.y,ma0.z,ma0.w, ma1.x,ma1.y,ma1.z,ma1.w,
                  ma2.x,ma2.y,ma2.z,ma2.w, ma3.x,ma3.y,ma3.z,ma3.w};
    const int me = l15 + 1;
    v16h A;
    #pragma unroll
    for (int i = 0; i < 16; ++i)
      A[i] = (_Float16)((li[i] * mi[i] == me) ? 1.0f : 0.0f);

    // ---- B fragment (16-bit B 32x16): lane half h, col N=l15, K = 8h..8h+7, 16+8h..16+8h+7
    //      two ds_load_b128 from eT
    v8h b0 = *(const v8h*)&eT[wave][l15][8*half];
    v8h b1 = *(const v8h*)&eT[wave][l15][16 + 8*half];
    v16h Bm;
    #pragma unroll
    for (int i = 0; i < 8; ++i) { Bm[i] = b0[i]; Bm[8 + i] = b1[i]; }

    acc = __builtin_amdgcn_wmma_f32_16x16x32_f16(false, A, false, Bm,
                                                 (short)0, acc, false, false);
    // next iteration's stores must not overtake this WMMA's source reads;
    // the dscnt/asynccnt waits above already order LDS ops, and WMMA consumed
    // its operands from VGPRs, so nothing extra is needed here.
  }

  // C/D layout: VGPR r -> M = r + 8*half, N = l15
  float* muAcc = W + OFF_MU_ACC + b * 256;
  #pragma unroll
  for (int r = 0; r < 8; ++r)
    atomicAdd(&muAcc[(r + 8*half) * 16 + l15], acc[r]);
}

// ---------------- kernel 5: finalize mu, compute per-batch push & Kb ---------------
__global__ void k_disc_fin(float* W) {
  int t = threadIdx.x;
  if (t < 64) {
    int b = t >> 3, k = t & 7;
    float cntRaw = W[OFF_MU_ACC + b * 256 + k * 16 + 8];
    float cnt = fmaxf(cntRaw, 1.0f);
    #pragma unroll
    for (int c = 0; c < 8; ++c)
      W[OFF_MU + b * 64 + k * 8 + c] = W[OFF_MU_ACC + b * 256 + k * 16 + c] / cnt;
    W[OFF_CNT + b * 8 + k] = cntRaw;
  }
  __syncthreads();
  if (t < BB) {
    int b = t;
    bool valid[KMAXI]; float Kb = 0;
    #pragma unroll
    for (int k = 0; k < KMAXI; ++k) {
      valid[k] = W[OFF_CNT + b * 8 + k] > 0.5f;
      Kb += valid[k] ? 1.0f : 0.0f;
    }
    float sum = 0, np = 0;
    for (int i = 0; i < KMAXI; ++i)
      for (int j = i + 1; j < KMAXI; ++j)
        if (valid[i] && valid[j]) {
          float d2 = 0;
          #pragma unroll
          for (int c = 0; c < 8; ++c) {
            float dv = W[OFF_MU + b*64 + i*8 + c] - W[OFF_MU + b*64 + j*8 + c];
            d2 += dv * dv;
          }
          float pd = sqrtf(fmaxf(d2, 1e-12f));
          float r = fmaxf(3.0f - pd, 0.0f);   // 2*DD = 3.0
          sum += r * r; np += 1.0f;
        }
    W[OFF_PUSHB + b] = sum / fmaxf(np, 1.0f);
    W[OFF_KB + b]    = Kb;
  }
}

// ---------------- kernel 6: pull pass (per-pixel distance to its instance mu) ------
__global__ void k_disc_pull(const float* __restrict__ emb, const int* __restrict__ lab,
                            const int* __restrict__ msk, float* W) {
  __shared__ float pullS[KMAXI];
  __shared__ float muS[KMAXI][8];
  const int b = blockIdx.x / BPB, blk = blockIdx.x % BPB;
  if (threadIdx.x < KMAXI) pullS[threadIdx.x] = 0.0f;
  if (threadIdx.x < 64)
    muS[threadIdx.x >> 3][threadIdx.x & 7] = W[OFF_MU + b * 64 + threadIdx.x];
  __syncthreads();

  const float* eb = emb + (size_t)b * 8 * HW;
  const int*   lb = lab + (size_t)b * HW;
  const int*   mb = msk + (size_t)b * HW;
  for (int pix = blk * (int)blockDim.x + (int)threadIdx.x; pix < HW;
       pix += BPB * (int)blockDim.x) {
    int iv = lb[pix] * mb[pix];
    if (iv > 0 && iv <= KMAXI) {
      int k = iv - 1;
      float d2 = 0;
      #pragma unroll
      for (int c = 0; c < 8; ++c) {
        float dv = eb[(size_t)c * HW + pix] - muS[k][c];
        d2 += dv * dv;
      }
      float d = sqrtf(fmaxf(d2, 1e-12f));
      float r = fmaxf(d - 0.5f, 0.0f);        // DV = 0.5
      atomicAdd(&pullS[k], r * r);
    }
  }
  __syncthreads();
  if (threadIdx.x < KMAXI)
    atomicAdd(&W[OFF_PULL + b * 8 + threadIdx.x], pullS[threadIdx.x]);
}

// ---------------- kernel 7: final scalar assembly ----------------------------------
__global__ void k_final(const float* W, float* out) {
  if (threadIdx.x != 0 || blockIdx.x != 0) return;
  const float* S = W + OFF_SCAL;
  int n_pos = (int)(S[S_NPOS] + 0.5f);
  int n_av  = (int)(S[S_NNEGAV] + 0.5f);
  int n_neg = n_pos > 0 ? min(n_av, (int)(3.0f * (float)n_pos)) : 100;
  float cnt_at = S[S_CNTAT], need = S[S_NEEDAT];
  float loss_neg = S[S_SUMABOVE] + (cnt_at > 0.5f ? S[S_SUMAT] * (need / cnt_at) : 0.0f);
  float loss_pos = n_pos > 0 ? S[S_POSSUM] : 0.0f;
  float loss_tr  = (loss_pos + loss_neg) / (float)(n_pos + n_neg);

  float n1 = S[S_N1];
  float loss_tcl = n1 > 0.5f ? S[S_TCLSUM] / fmaxf(n1, 1.0f) : 0.0f;

  bool  has2 = S[S_HAS2] > 0.5f;
  float n2   = fmaxf(S[S_N2], 1.0f);
  float loss_radii = has2 ? S[S_RADSUM] / n2 : 0.0f;
  float loss_sin   = has2 ? S[S_SINSUM] / n2 : 0.0f;
  float loss_cos   = has2 ? S[S_COSSUM] / n2 : 0.0f;

  float pullT = 0, pushT = 0, nv = 0;
  for (int b = 0; b < BB; ++b) {
    float Kb = W[OFF_KB + b];
    if (Kb > 0.5f) {
      nv += 1.0f;
      float pb = 0;
      for (int k = 0; k < KMAXI; ++k) {
        float cnt = fmaxf(W[OFF_CNT + b * 8 + k], 1.0f);
        pb += W[OFF_PULL + b * 8 + k] / cnt;
      }
      pullT += pb / fmaxf(Kb, 1.0f);
      pushT += W[OFF_PUSHB + b];
    }
  }
  float pull = nv > 0.5f ? pullT / nv : 0.0f;
  float push = nv > 0.5f ? pushT / nv : 0.0f;

  float total = loss_tr + loss_tcl + loss_radii + loss_sin + loss_cos + 0.1f * (pull + push);
  out[0] = total;      out[1] = loss_tr;  out[2] = loss_tcl; out[3] = loss_radii;
  out[4] = loss_sin;   out[5] = loss_cos; out[6] = pull;     out[7] = push;
}

extern "C" void kernel_launch(void* const* d_in, const int* in_sizes, int n_in,
                              void* d_out, int out_size, void* d_ws, size_t ws_size,
                              hipStream_t stream) {
  const float* inp   = (const float*)d_in[0];
  const float* emb   = (const float*)d_in[1];
  const int*   trm   = (const int*)d_in[2];
  const int*   tclm  = (const int*)d_in[3];
  const float* sm    = (const float*)d_in[4];
  const float* cm    = (const float*)d_in[5];
  const float* rm    = (const float*)d_in[6];
  const int*   tm    = (const int*)d_in[7];
  const int*   ilab  = (const int*)d_in[8];
  float* W   = (float*)d_ws;
  float* out = (float*)d_out;

  k_zero<<<32, 256, 0, stream>>>(W, WS_FLOATS);
  k_pass1<<<1024, 256, 0, stream>>>(inp, trm, tclm, sm, cm, rm, tm, W);
  k_select<<<1, 32, 0, stream>>>(W);
  k_pass1c<<<1024, 256, 0, stream>>>(inp, trm, tm, W);
  k_disc_mu<<<BB * BPB, 256, 0, stream>>>(emb, ilab, tclm, W);
  k_disc_fin<<<1, 64, 0, stream>>>(W);
  k_disc_pull<<<BB * BPB, 256, 0, stream>>>(emb, ilab, tclm, W);
  k_final<<<1, 32, 0, stream>>>(W, out);
}
